// HLFormerBlock_81509889344241
// MI455X (gfx1250) — matrix operand order repaired and checked by reading, not compiled, not run on hardware
//
#include <hip/hip_runtime.h>

// ---------------------------------------------------------------------------
// HLFormer block for MI455X (gfx1250): bf16 WMMA GEMMs + fused attention.
// Round 2: weights pre-transposed at conversion; GEMM B-tile staged with
// CDNA5 async global->LDS copies (global_load_async_to_lds_b128 + asynccnt).
// ---------------------------------------------------------------------------

constexpr int  kN   = 64;
constexpr int  kL   = 256;
constexpr int  kD   = 768;
constexpr int  kH   = 12;
constexpr int  kB   = 4;
constexpr int  kDFF = 3072;
constexpr int  kDH  = 64;
constexpr long kNLD = (long)kN * kL * kD;

typedef __attribute__((ext_vector_type(16))) __bf16 v16bf;
typedef __attribute__((ext_vector_type(8)))  float  v8f;

union Frag { v16bf v; unsigned short u[16]; };

__device__ __forceinline__ unsigned short f2bf(float f) {
  unsigned u = __builtin_bit_cast(unsigned, f);
  u += 0x7FFFu + ((u >> 16) & 1u);              // round-to-nearest-even
  return (unsigned short)(u >> 16);
}

// CDNA5 async global -> LDS copy, 16B per lane, tracked by ASYNCcnt.
__device__ __forceinline__ void async_ld_b128(unsigned lds_off, const void* gptr) {
  asm volatile("global_load_async_to_lds_b128 %0, %1, off"
               :: "v"(lds_off), "v"(gptr)
               : "memory");
}
__device__ __forceinline__ void wait_async0() {
  asm volatile("s_wait_asynccnt 0x0" ::: "memory");
}

// ---------------------------------------------------------------------------
// Weight conversion with transpose: dst[n][k] (bf16) = src[k][n] (f32).
// 32x32 tiles through LDS; K, N multiples of 32 for every weight here.
// ---------------------------------------------------------------------------
__global__ void cvt_t_kernel(const float* __restrict__ src,
                             unsigned short* __restrict__ dst, int K, int N) {
  __shared__ float tile[32][33];
  const int tx = threadIdx.x, ty = threadIdx.y;       // 32 x 8
  const int k0 = blockIdx.y * 32, n0 = blockIdx.x * 32;
  for (int j = 0; j < 4; j++)
    tile[ty + j * 8][tx] = src[(size_t)(k0 + ty + j * 8) * N + n0 + tx];
  __syncthreads();
  for (int j = 0; j < 4; j++)
    dst[(size_t)(n0 + ty + j * 8) * K + k0 + tx] = f2bf(tile[tx][ty + j * 8]);
}

// ---------------------------------------------------------------------------
// Gaussian masks: gmm[widx][q][x] = exp(u - rowmax(u)),
// u = -(x/255 - q/256)^2 / (2 (wid/9)^2).  (C/width factor cancels in max-norm)
// ---------------------------------------------------------------------------
__global__ void gauss_kernel(float* __restrict__ gmm) {
  const int widx = blockIdx.x >> 8;
  const int q    = blockIdx.x & 255;
  const float wids[3] = {0.25f, 0.5f, 1.0f};
  const float width  = wids[widx] / 9.0f;
  const float center = (float)q / 256.0f;
  const float x      = (float)threadIdx.x / 255.0f;
  const float dlt    = x - center;
  const float u      = -(dlt * dlt) / (2.0f * width * width);
  __shared__ float red[8];
  float m = u;
  for (int o = 16; o > 0; o >>= 1) m = fmaxf(m, __shfl_xor(m, o, 32));
  const int lane = threadIdx.x & 31, wid = threadIdx.x >> 5;
  if (lane == 0) red[wid] = m;
  __syncthreads();
  float bm = red[0];
  for (int i = 1; i < 8; i++) bm = fmaxf(bm, red[i]);
  gmm[(size_t)widx * kL * kL + (size_t)q * kL + threadIdx.x] = expf(u - bm);
}

// ---------------------------------------------------------------------------
// Generic GEMM: C[MxNc] = act(A[MxK](f32) @ W[KxNc] + bias), with the weight
// supplied pre-transposed as Wt[Nc][K] in bf16.
// Block tile 128x128, BK=32, 256 threads = 8 waves (4 in M x 2 in N),
// each wave: 2x4 sub-tiles of v_wmma_f32_16x16x32_bf16.
// A staged f32->bf16 through VALU; B staged by async global->LDS b128 copies.
// ---------------------------------------------------------------------------
__launch_bounds__(256)
__global__ void gemm_bf16(const float* __restrict__ A,
                          const unsigned short* __restrict__ Wt,
                          const float* __restrict__ bias,
                          float* __restrict__ C,
                          int M, int Nc, int K, int relu) {
  __shared__ unsigned short As[128 * 40];   // [m][k], pad to 40
  __shared__ unsigned short Bs[128 * 40];   // [n][k], pad to 40

  const int tid    = threadIdx.x;
  const int lane   = tid & 31, wid = tid >> 5;
  const int wave_m = wid & 3,  wave_n = wid >> 2;
  const int half   = lane >> 4, l16 = lane & 15;
  const int row0   = blockIdx.y * 128, col0 = blockIdx.x * 128;

  v8f acc[2][4];
  for (int i = 0; i < 2; i++)
    for (int j = 0; j < 4; j++)
      for (int e = 0; e < 8; e++) acc[i][j][e] = 0.0f;

  const int ar  = tid >> 3;         // 0..31
  const int ac4 = (tid & 7) * 4;    // 0..28

  for (int kk0 = 0; kk0 < K; kk0 += 32) {
    // ---- stage B: async global->LDS, 2 x 16B per thread (8KB tile) ----
    for (int p = 0; p < 2; p++) {
      const int t  = tid + p * 256;
      const int n  = t >> 2;        // 0..127
      const int ch = t & 3;         // 16B chunk within the 64B row
      const unsigned loff = (unsigned)(size_t)&Bs[n * 40 + ch * 8];
      const unsigned short* g = Wt + (size_t)(col0 + n) * K + kk0 + ch * 8;
      async_ld_b128(loff, g);
      if (kk0 + 32 < K) __builtin_prefetch(g + 32, 0, 1);
    }
    // ---- stage A (f32 -> bf16) ----
    for (int p = 0; p < 4; p++) {
      const int rl = p * 32 + ar;
      const int gr = row0 + rl;
      float4 v = make_float4(0.f, 0.f, 0.f, 0.f);
      if (gr < M) {
        v = *(const float4*)(A + (size_t)gr * K + kk0 + ac4);
        if (kk0 + 32 < K)
          __builtin_prefetch(A + (size_t)gr * K + kk0 + 32 + ac4, 0, 1);
      }
      unsigned short* d = &As[rl * 40 + ac4];
      d[0] = f2bf(v.x); d[1] = f2bf(v.y); d[2] = f2bf(v.z); d[3] = f2bf(v.w);
    }
    wait_async0();
    __syncthreads();

    Frag fa[2], fb[4];
    for (int sm = 0; sm < 2; sm++) {
      const int mloc = wave_m * 32 + sm * 16 + l16;
      for (int j = 0; j < 8; j++) {
        const int k0 = 2 * j + 16 * half;
        const unsigned pv = *(const unsigned*)&As[mloc * 40 + k0];
        fa[sm].u[2 * j] = (unsigned short)pv;
        fa[sm].u[2 * j + 1] = (unsigned short)(pv >> 16);
      }
    }
    for (int sn = 0; sn < 4; sn++) {
      const int nloc = wave_n * 64 + sn * 16 + l16;
      for (int j = 0; j < 8; j++) {
        const int k0 = 2 * j + 16 * half;
        const unsigned pv = *(const unsigned*)&Bs[nloc * 40 + k0];
        fb[sn].u[2 * j] = (unsigned short)pv;
        fb[sn].u[2 * j + 1] = (unsigned short)(pv >> 16);
      }
    }
    for (int sm = 0; sm < 2; sm++)
      for (int sn = 0; sn < 4; sn++)
        acc[sm][sn] = __builtin_amdgcn_wmma_f32_16x16x32_bf16(
            false, fa[sm].v, false, fb[sn].v, (short)0, acc[sm][sn], false, false);
    __syncthreads();
  }

  for (int sm = 0; sm < 2; sm++)
    for (int sn = 0; sn < 4; sn++)
      for (int r = 0; r < 8; r++) {
        const int m = row0 + wave_m * 32 + sm * 16 + r + 8 * half;
        const int n = col0 + wave_n * 64 + sn * 16 + l16;
        if (m < M && n < Nc) {
          float v = acc[sm][sn][r] + (bias ? bias[n] : 0.0f);
          if (relu) v = fmaxf(v, 0.0f);
          C[(size_t)m * Nc + n] = v;
        }
      }
}

// ---------------------------------------------------------------------------
// Fused attention for one (n, h, 64-query tile):
//   S = (Q K^T)/8 (* gmm), softmax rows, O = P V.   All via bf16 WMMA.
// LDS: Qs 9.2KB, K/V 36.9KB (reused), S strip 66.6KB  (~112KB total)
// ---------------------------------------------------------------------------
__launch_bounds__(256)
__global__ void attn_kernel(const float* __restrict__ qb,
                            const float* __restrict__ kb,
                            const float* __restrict__ vb,
                            const float* __restrict__ gmm,   // null => no mask
                            float* __restrict__ ob) {
  __shared__ unsigned short Qs[64 * 72];     // [q][d]
  __shared__ unsigned short KVs[256 * 72];   // K: [kk][d]; later V^T: [d][kk] (stride 264)
  __shared__ float Ss[64 * 260];             // scores / probabilities

  const int nh   = blockIdx.x;
  const int n    = nh / kH, h = nh % kH;
  const int q0   = blockIdx.y * 64;
  const int tid  = threadIdx.x, lane = tid & 31, wid = tid >> 5;
  const int half = lane >> 4, l16 = lane & 15;

  // ---- stage Q (64x64) and K (256x64) as bf16 ----
  {
    const int r = tid >> 4, c4 = (tid & 15) * 4;
    for (int p = 0; p < 4; p++) {
      const int rl = p * 16 + r;
      float4 v = *(const float4*)(qb + (size_t)(n * kL + q0 + rl) * kD + h * kDH + c4);
      unsigned short* d = &Qs[rl * 72 + c4];
      d[0] = f2bf(v.x); d[1] = f2bf(v.y); d[2] = f2bf(v.z); d[3] = f2bf(v.w);
    }
    for (int p = 0; p < 16; p++) {
      const int rl = p * 16 + r;
      float4 v = *(const float4*)(kb + (size_t)(n * kL + rl) * kD + h * kDH + c4);
      unsigned short* d = &KVs[rl * 72 + c4];
      d[0] = f2bf(v.x); d[1] = f2bf(v.y); d[2] = f2bf(v.z); d[3] = f2bf(v.w);
    }
  }
  __syncthreads();

  // ---- S = Q @ K^T : wave handles 16 rows x 128 cols (8 sub-tiles) ----
  {
    const int m_sub = wid & 3;
    const int nb    = (wid >> 2) * 8;
    v8f acc[8];
    for (int t = 0; t < 8; t++)
      for (int e = 0; e < 8; e++) acc[t][e] = 0.0f;
    for (int c0 = 0; c0 < kDH; c0 += 32) {
      Frag fa;
      const int mloc = m_sub * 16 + l16;
      for (int j = 0; j < 8; j++) {
        const int k0 = 2 * j + 16 * half;
        const unsigned pv = *(const unsigned*)&Qs[mloc * 72 + c0 + k0];
        fa.u[2 * j] = (unsigned short)pv; fa.u[2 * j + 1] = (unsigned short)(pv >> 16);
      }
      for (int t = 0; t < 8; t++) {
        Frag fbq;
        const int nloc = (nb + t) * 16 + l16;   // kk column
        for (int j = 0; j < 8; j++) {
          const int k0 = 2 * j + 16 * half;
          const unsigned pv = *(const unsigned*)&KVs[nloc * 72 + c0 + k0];
          fbq.u[2 * j] = (unsigned short)pv; fbq.u[2 * j + 1] = (unsigned short)(pv >> 16);
        }
        acc[t] = __builtin_amdgcn_wmma_f32_16x16x32_bf16(
            false, fa.v, false, fbq.v, (short)0, acc[t], false, false);
      }
    }
    for (int t = 0; t < 8; t++)
      for (int r = 0; r < 8; r++)
        Ss[(m_sub * 16 + r + 8 * half) * 260 + (nb + t) * 16 + l16] = acc[t][r];
  }
  __syncthreads();

  // ---- scale, mask, softmax (wave handles 8 rows) ----
  for (int rr = 0; rr < 8; rr++) {
    const int row = wid * 8 + rr;
    float v[8];
    float m = -1e30f;
    for (int i = 0; i < 8; i++) {
      const int col = lane + i * 32;
      float s = Ss[row * 260 + col] * 0.125f;
      if (gmm) s *= gmm[(size_t)(q0 + row) * kL + col];
      v[i] = s; m = fmaxf(m, s);
    }
    for (int o = 16; o > 0; o >>= 1) m = fmaxf(m, __shfl_xor(m, o, 32));
    float sum = 0.f;
    for (int i = 0; i < 8; i++) { v[i] = expf(v[i] - m); sum += v[i]; }
    for (int o = 16; o > 0; o >>= 1) sum += __shfl_xor(sum, o, 32);
    const float inv = 1.0f / sum;
    for (int i = 0; i < 8; i++) Ss[row * 260 + lane + i * 32] = v[i] * inv;
  }
  __syncthreads();

  // ---- stage V transposed: Vt[d][kk] (stride 264), reusing KVs ----
  {
    const int r = tid >> 4, c4 = (tid & 15) * 4;
    for (int p = 0; p < 16; p++) {
      const int kk = p * 16 + r;
      float4 v = *(const float4*)(vb + (size_t)(n * kL + kk) * kD + h * kDH + c4);
      KVs[(c4 + 0) * 264 + kk] = f2bf(v.x);
      KVs[(c4 + 1) * 264 + kk] = f2bf(v.y);
      KVs[(c4 + 2) * 264 + kk] = f2bf(v.z);
      KVs[(c4 + 3) * 264 + kk] = f2bf(v.w);
    }
  }
  __syncthreads();

  // ---- O = P @ V : 16 sub-tiles, wave handles tiles 2w, 2w+1 ----
  {
    const int t0     = wid * 2;
    const int m_sub  = t0 >> 2;
    const int n_sub0 = t0 & 3;
    v8f acc0, acc1;
    for (int e = 0; e < 8; e++) { acc0[e] = 0.f; acc1[e] = 0.f; }
    for (int c0 = 0; c0 < kL; c0 += 32) {
      Frag fa;
      const int mloc = m_sub * 16 + l16;
      for (int j = 0; j < 8; j++) {
        const int k0 = 2 * j + 16 * half;
        fa.u[2 * j]     = f2bf(Ss[mloc * 260 + c0 + k0]);
        fa.u[2 * j + 1] = f2bf(Ss[mloc * 260 + c0 + k0 + 1]);
      }
      Frag fb0, fb1;
      for (int j = 0; j < 8; j++) {
        const int k0 = c0 + 2 * j + 16 * half;
        const unsigned p0 = *(const unsigned*)&KVs[(n_sub0 * 16 + l16) * 264 + k0];
        const unsigned p1 = *(const unsigned*)&KVs[((n_sub0 + 1) * 16 + l16) * 264 + k0];
        fb0.u[2 * j] = (unsigned short)p0; fb0.u[2 * j + 1] = (unsigned short)(p0 >> 16);
        fb1.u[2 * j] = (unsigned short)p1; fb1.u[2 * j + 1] = (unsigned short)(p1 >> 16);
      }
      acc0 = __builtin_amdgcn_wmma_f32_16x16x32_bf16(
          false, fa.v, false, fb0.v, (short)0, acc0, false, false);
      acc1 = __builtin_amdgcn_wmma_f32_16x16x32_bf16(
          false, fa.v, false, fb1.v, (short)0, acc1, false, false);
    }
    for (int r = 0; r < 8; r++) {
      const size_t row = (size_t)(n * kL + q0 + m_sub * 16 + r + 8 * half) * kD + h * kDH;
      ob[row + n_sub0 * 16 + l16]       = acc0[r];
      ob[row + (n_sub0 + 1) * 16 + l16] = acc1[r];
    }
  }
}

// ---------------------------------------------------------------------------
// out[row] = LN(a[row] + b[row]; g, bt)  — one block per row of 768
// ---------------------------------------------------------------------------
__global__ void add_ln_kernel(const float* __restrict__ a, const float* __restrict__ b,
                              const float* __restrict__ g, const float* __restrict__ bt,
                              float* __restrict__ out, long out_stride) {
  const int row = blockIdx.x;
  const int tid = threadIdx.x;
  __shared__ float sA[8], sB[8];
  float x[3], s = 0.f, s2 = 0.f;
  for (int i = 0; i < 3; i++) {
    const int c = tid + i * 256;
    const float v = a[(size_t)row * kD + c] + b[(size_t)row * kD + c];
    x[i] = v; s += v; s2 += v * v;
  }
  for (int o = 16; o > 0; o >>= 1) { s += __shfl_xor(s, o, 32); s2 += __shfl_xor(s2, o, 32); }
  const int lane = tid & 31, wid = tid >> 5;
  if (lane == 0) { sA[wid] = s; sB[wid] = s2; }
  __syncthreads();
  s = 0.f; s2 = 0.f;
  for (int i = 0; i < 8; i++) { s += sA[i]; s2 += sB[i]; }
  const float mean = s / kD;
  const float var  = s2 / kD - mean * mean;
  const float inv  = rsqrtf(var + 1e-5f);
  for (int i = 0; i < 3; i++) {
    const int c = tid + i * 256;
    out[(size_t)row * out_stride + c] = g[c] * (x[i] - mean) * inv + bt[c];
  }
}

// ---------------------------------------------------------------------------
// wt[n,d] = mean over (block, l) of oo
// ---------------------------------------------------------------------------
__global__ void wt_kernel(const float* __restrict__ oo, float* __restrict__ wt) {
  const int idx = blockIdx.x * 256 + threadIdx.x;
  if (idx >= kN * kD) return;
  const int n = idx / kD, d = idx % kD;
  float s = 0.f;
  for (int i = 0; i < kB; i++)
    for (int l = 0; l < kL; l++)
      s += oo[(size_t)i * kNLD + (size_t)(n * kL + l) * kD + d];
  wt[idx] = s * (1.0f / (kB * kL));
}

// ---------------------------------------------------------------------------
// Cross attention (lq = 1): one block per n.  fp32 VALU (tiny).
// ---------------------------------------------------------------------------
__launch_bounds__(256)
__global__ void cross_attn_kernel(const float* __restrict__ qc, const float* __restrict__ kc,
                                  const float* __restrict__ vc, float* __restrict__ ab) {
  const int n = blockIdx.x;
  const int tid = threadIdx.x;
  __shared__ float sc[kH * kL];
  for (int e = tid; e < kH * kL; e += 256) {
    const int h = e >> 8, kk = e & 255;
    const float* qp = qc + (size_t)n * kD + h * kDH;
    const float* kp = kc + (size_t)(n * kL + kk) * kD + h * kDH;
    float d = 0.f;
    for (int i = 0; i < kDH; i++) d += qp[i] * kp[i];
    sc[e] = d * 0.125f;
  }
  __syncthreads();
  const int lane = tid & 31, wid = tid >> 5;
  for (int h = wid; h < kH; h += 8) {
    float v[8], m = -1e30f;
    for (int i = 0; i < 8; i++) { v[i] = sc[h * 256 + lane + i * 32]; m = fmaxf(m, v[i]); }
    for (int o = 16; o > 0; o >>= 1) m = fmaxf(m, __shfl_xor(m, o, 32));
    float s = 0.f;
    for (int i = 0; i < 8; i++) { v[i] = expf(v[i] - m); s += v[i]; }
    for (int o = 16; o > 0; o >>= 1) s += __shfl_xor(s, o, 32);
    const float inv = 1.f / s;
    for (int i = 0; i < 8; i++) sc[h * 256 + lane + i * 32] = v[i] * inv;
  }
  __syncthreads();
  for (int i = 0; i < 3; i++) {
    const int d = tid + i * 256;
    const int h = d / kDH;
    float s = 0.f;
    for (int kk = 0; kk < kL; kk++)
      s += sc[h * 256 + kk] * vc[(size_t)(n * kL + kk) * kD + d];
    ab[(size_t)n * kD + d] = s;
  }
}

// ---------------------------------------------------------------------------
// Final: out[n,l,:] = sum_i softmax_i(logits[n,i,l]/0.1) * oo_i[n,l,:]
// ---------------------------------------------------------------------------
__global__ void combine_kernel(const float* __restrict__ oo,
                               const float* __restrict__ logits,
                               float* __restrict__ out) {
  const int bid = blockIdx.x;           // n*256 + l
  const int n = bid >> 8, l = bid & 255;
  const int tid = threadIdx.x;
  float lg[kB], m = -1e30f;
  for (int i = 0; i < kB; i++) {
    lg[i] = logits[(size_t)(n * kB + i) * kL + l] * 10.0f;
    m = fmaxf(m, lg[i]);
  }
  float s = 0.f;
  for (int i = 0; i < kB; i++) { lg[i] = expf(lg[i] - m); s += lg[i]; }
  const float inv = 1.f / s;
  for (int j = 0; j < 3; j++) {
    const int c = tid + j * 256;
    const size_t base = (size_t)bid * kD + c;
    float acc = 0.f;
    for (int i = 0; i < kB; i++) acc += lg[i] * inv * oo[(size_t)i * kNLD + base];
    out[base] = acc;
  }
}

// ---------------------------------------------------------------------------
// Host orchestration
// ---------------------------------------------------------------------------
extern "C" void kernel_launch(void* const* d_in, const int* in_sizes, int n_in,
                              void* d_out, int out_size, void* d_ws, size_t ws_size,
                              hipStream_t stream) {
  (void)in_sizes; (void)n_in; (void)out_size; (void)ws_size;

  const float* x    = (const float*)d_in[0];
  const float* eWq  = (const float*)d_in[1];
  const float* ebq  = (const float*)d_in[2];
  const float* eWk  = (const float*)d_in[3];
  const float* ebk  = (const float*)d_in[4];
  const float* eWv  = (const float*)d_in[5];
  const float* ebv  = (const float*)d_in[6];
  const float* eW1  = (const float*)d_in[7];
  const float* eb1  = (const float*)d_in[8];
  const float* eW2  = (const float*)d_in[9];
  const float* eb2  = (const float*)d_in[10];
  const float* eg1  = (const float*)d_in[11];
  const float* ebt1 = (const float*)d_in[12];
  const float* eg2  = (const float*)d_in[13];
  const float* ebt2 = (const float*)d_in[14];
  const float* cWq  = (const float*)d_in[15];
  const float* cbq  = (const float*)d_in[16];
  const float* cWk  = (const float*)d_in[17];
  const float* cbk  = (const float*)d_in[18];
  const float* cWv  = (const float*)d_in[19];
  const float* cbv  = (const float*)d_in[20];
  const float* cW1  = (const float*)d_in[21];
  const float* cb1  = (const float*)d_in[22];
  const float* cW2  = (const float*)d_in[23];
  const float* cb2  = (const float*)d_in[24];
  const float* cg1  = (const float*)d_in[25];
  const float* cbt1 = (const float*)d_in[26];
  const float* cg2  = (const float*)d_in[27];
  const float* cbt2 = (const float*)d_in[28];
  const float* W1   = (const float*)d_in[29];
  const float* b1   = (const float*)d_in[30];
  const float* W2   = (const float*)d_in[31];
  const float* b2   = (const float*)d_in[32];

  char* ws = (char*)d_ws;
  size_t off = 0;
  auto alc = [&](size_t bytes) -> char* {
    char* p = ws + off;
    off = (off + bytes + 255) & ~(size_t)255;
    return p;
  };

  // bf16 weight pool (all stored TRANSPOSED: [n][k])
  unsigned short* bWq  = (unsigned short*)alc((size_t)kB * kD * kD * 2);
  unsigned short* bWk  = (unsigned short*)alc((size_t)kB * kD * kD * 2);
  unsigned short* bWv  = (unsigned short*)alc((size_t)kB * kD * kD * 2);
  unsigned short* bW1  = (unsigned short*)alc((size_t)kB * kD * kDFF * 2);
  unsigned short* bW2  = (unsigned short*)alc((size_t)kB * kDFF * kD * 2);
  unsigned short* bcWq = (unsigned short*)alc((size_t)kD * kD * 2);
  unsigned short* bcWk = (unsigned short*)alc((size_t)kD * kD * 2);
  unsigned short* bcWv = (unsigned short*)alc((size_t)kD * kD * 2);
  unsigned short* bcW1 = (unsigned short*)alc((size_t)kD * kD * 2);
  unsigned short* bcW2 = (unsigned short*)alc((size_t)kD * kD * 2);
  unsigned short* bWf1 = (unsigned short*)alc((size_t)kD * kD * 2);
  unsigned short* bWf2 = (unsigned short*)alc((size_t)kD * kL * 2);
  float* gmm  = (float*)alc((size_t)3 * kL * kL * 4);
  // region A: q/k/v (3*kNLD) aliased with FFN hidden (N*L*DFF)
  float* regA = (float*)alc((size_t)kN * kL * kDFF * 4);
  float* qbuf = regA;
  float* kbuf = regA + kNLD;
  float* vbuf = regA + 2 * kNLD;
  float* ffnh = regA;
  // region B: attention out / FFN t2
  float* regB = (float*)alc((size_t)kNLD * 4);
  float* hbuf = (float*)alc((size_t)kNLD * 4);
  float* oo   = (float*)alc((size_t)kB * kNLD * 4);
  float* wt   = (float*)alc((size_t)kN * kD * 4);
  float* qc   = (float*)alc((size_t)kN * kD * 4);
  float* ab   = (float*)alc((size_t)kN * kD * 4);
  float* q2   = (float*)alc((size_t)kN * kD * 4);
  float* ct1  = (float*)alc((size_t)kN * kD * 4);
  float* ct2  = (float*)alc((size_t)kN * kD * 4);
  float* wsb  = (float*)alc((size_t)kN * kB * kD * 4);
  float* r1   = (float*)alc((size_t)kN * kB * kD * 4);
  float* logi = (float*)alc((size_t)kN * kB * kL * 4);

  // transpose-convert: src f32 [K][N] -> dst bf16 [N][K]
  auto cvtT = [&](const float* s, unsigned short* d, int K, int N) {
    dim3 g(N / 32, K / 32);
    cvt_t_kernel<<<g, dim3(32, 8), 0, stream>>>(s, d, K, N);
  };
  auto gemm = [&](const float* A, const unsigned short* Wt, const float* bias,
                  float* C, int M, int Nc, int K, int relu) {
    dim3 g((Nc + 127) / 128, (M + 127) / 128);
    gemm_bf16<<<g, 256, 0, stream>>>(A, Wt, bias, C, M, Nc, K, relu);
  };

  // --- weight conversions (with transpose) + gaussian masks ---
  for (int i = 0; i < kB; i++) {
    cvtT(eWq + (size_t)i * kD * kD, bWq + (size_t)i * kD * kD, kD, kD);
    cvtT(eWk + (size_t)i * kD * kD, bWk + (size_t)i * kD * kD, kD, kD);
    cvtT(eWv + (size_t)i * kD * kD, bWv + (size_t)i * kD * kD, kD, kD);
    cvtT(eW1 + (size_t)i * kD * kDFF, bW1 + (size_t)i * kD * kDFF, kD, kDFF);
    cvtT(eW2 + (size_t)i * kDFF * kD, bW2 + (size_t)i * kDFF * kD, kDFF, kD);
  }
  cvtT(cWq, bcWq, kD, kD);
  cvtT(cWk, bcWk, kD, kD);
  cvtT(cWv, bcWv, kD, kD);
  cvtT(cW1, bcW1, kD, kD);
  cvtT(cW2, bcW2, kD, kD);
  cvtT(W1,  bWf1, kD, kD);
  cvtT(W2,  bWf2, kD, kL);
  gauss_kernel<<<3 * kL, 256, 0, stream>>>(gmm);

  const int ML = kN * kL;  // 16384

  // --- Euclidean attention blocks ---
  for (int i = 0; i < kB; i++) {
    gemm(x, bWq + (size_t)i * kD * kD, ebq + i * kD, qbuf, ML, kD, kD, 0);
    gemm(x, bWk + (size_t)i * kD * kD, ebk + i * kD, kbuf, ML, kD, kD, 0);
    gemm(x, bWv + (size_t)i * kD * kD, ebv + i * kD, vbuf, ML, kD, kD, 0);
    attn_kernel<<<dim3(kN * kH, kL / 64), 256, 0, stream>>>(
        qbuf, kbuf, vbuf, (i == 0) ? nullptr : (gmm + (size_t)(i - 1) * kL * kL), regB);
    add_ln_kernel<<<ML, 256, 0, stream>>>(x, regB, eg1 + i * kD, ebt1 + i * kD, hbuf, kD);
    gemm(hbuf, bW1 + (size_t)i * kD * kDFF, eb1 + i * kDFF, ffnh, ML, kDFF, kD, 1);
    gemm(ffnh, bW2 + (size_t)i * kDFF * kD, eb2 + i * kD, regB, ML, kD, kDFF, 0);
    add_ln_kernel<<<ML, 256, 0, stream>>>(hbuf, regB, eg2 + i * kD, ebt2 + i * kD,
                                          oo + (size_t)i * kNLD, kD);
  }

  // --- weight token ---
  wt_kernel<<<(kN * kD + 255) / 256, 256, 0, stream>>>(oo, wt);
  gemm(wt, bcWq, cbq, qc, kN, kD, kD, 0);

  // --- cross-attention weighting per block ---
  for (int i = 0; i < kB; i++) {
    gemm(oo + (size_t)i * kNLD, bcWk, cbk, regA, ML, kD, kD, 0);           // kc
    gemm(oo + (size_t)i * kNLD, bcWv, cbv, regA + kNLD, ML, kD, kD, 0);    // vc
    cross_attn_kernel<<<kN, 256, 0, stream>>>(qc, regA, regA + kNLD, ab);
    add_ln_kernel<<<kN, 256, 0, stream>>>(wt, ab, cg1, cbt1, q2, kD);
    gemm(q2, bcW1, cb1, ct1, kN, kD, kD, 1);
    gemm(ct1, bcW2, cb2, ct2, kN, kD, kD, 0);
    add_ln_kernel<<<kN, 256, 0, stream>>>(q2, ct2, cg2, cbt2, wsb + (size_t)i * kD,
                                          (long)kB * kD);
  }

  // --- final weighting MLP + softmax combine ---
  gemm(wsb, bWf1, b1, r1, kN * kB, kD, kD, 1);
  gemm(r1, bWf2, b2, logi, kN * kB, kL, kD, 0);
  combine_kernel<<<kN * kL, 256, 0, stream>>>(oo, logi, (float*)d_out);
}
